// QLSTM_65481071409234
// MI455X (gfx1250) — matrix operand order repaired and checked by reading, not compiled, not run on hardware
//
#include <hip/hip_runtime.h>
#include <hip/hip_bf16.h>
#include <cstdint>

typedef __attribute__((ext_vector_type(16))) __bf16 v16bf;
typedef __attribute__((ext_vector_type(8)))  __bf16 v8bf;
typedef __attribute__((ext_vector_type(4)))  __bf16 v4bf;
typedef __attribute__((ext_vector_type(8)))  float  v8f;
typedef __attribute__((ext_vector_type(4)))  unsigned int v4u;
typedef __attribute__((ext_vector_type(8)))  int    v8i;
typedef __attribute__((ext_vector_type(4)))  int    v4i;

#define TSTEPS 512
#define BATCH  64
#define INDIM  512
#define HID    512
#define COMB   1024
#define NWG    32          // each owns 16 hidden units -> 64 gate columns
#define NTHR   512         // 16 waves

// LDS layout (dynamic):
//   sW : 64 rows x 1024 K bf16  ([gate*16+n][k], contiguous 128 KB image)
//   sC : 64 rows x 1024 K bf16  (comb = [x_t | h])
//   sG : 4 x 64 x 16 float gate pre-activations
//   sB : 64 float biases
#define LDS_BYTES (131072 + 131072 + 16384 + 256)

// ws layout: [0,64) barrier | [64, +4MB) Wbf | then 128KB Hbuf x2 | then 32MB Xbf
#define WS_WOFF   64
#define WS_HOFF   (WS_WOFF + (size_t)4 * HID * COMB * 2)
#define WS_XOFF   (WS_HOFF + (size_t)2 * BATCH * HID * 2)
#define WS_NEED   (WS_XOFF + (size_t)TSTEPS * BATCH * INDIM * 2)

#if defined(__has_builtin)
#if __has_builtin(__builtin_amdgcn_tensor_load_to_lds)
#define HAVE_TDM 1
#endif
#endif

__device__ __forceinline__ void async_g2l_b128(unsigned lds_off, const void* gsrc) {
  asm volatile("global_load_async_to_lds_b128 %0, %1, off"
               :: "v"(lds_off), "v"(gsrc) : "memory");
}

// Fast saturating nonlinearities: v_exp_f32 + v_rcp_f32 (no IEEE div sequence).
__device__ __forceinline__ float sig_f(float x) {
  return __builtin_amdgcn_rcpf(1.f + __expf(-x));
}
__device__ __forceinline__ float tanh_f(float x) {
  float e = __expf(2.f * x);
  return (e - 1.f) * __builtin_amdgcn_rcpf(e + 1.f);
}

// Weights -> bf16, reorganized so each workgroup's slice is one contiguous
// 128 KB image: dst[((wg*4 + gate)*16 + r)*1024 + k], wg = n>>4, r = n&15.
__global__ void convert_weights_bf16(const float* __restrict__ Wf,
                                     const float* __restrict__ Wi,
                                     const float* __restrict__ Wg,
                                     const float* __restrict__ Wo,
                                     __bf16* __restrict__ dst) {
  int idx = blockIdx.x * blockDim.x + threadIdx.x;   // 4*512*1024 total
  if (idx >= 4 * HID * COMB) return;
  int gate = idx >> 19;
  int rem  = idx & ((1 << 19) - 1);
  int n = rem >> 10, k = rem & 1023;
  const float* src = (gate == 0) ? Wf : (gate == 1) ? Wi : (gate == 2) ? Wg : Wo;
  int wgI = n >> 4, r = n & 15;
  dst[(((size_t)(wgI * 4 + gate) * 16 + r) << 10) | k] = (__bf16)src[(size_t)n * COMB + k];
}

__global__ void convert_x_bf16(const float* __restrict__ X, __bf16* __restrict__ dst) {
  size_t i4 = (size_t)blockIdx.x * blockDim.x + threadIdx.x;  // float4 index
  if (i4 >= (size_t)TSTEPS * BATCH * INDIM / 4) return;
  float4 v = ((const float4*)X)[i4];
  v4bf hv = { (__bf16)v.x, (__bf16)v.y, (__bf16)v.z, (__bf16)v.w };
  ((v4bf*)dst)[i4] = hv;
}

__global__ void __launch_bounds__(NTHR, 1)
lstm_persistent(const float* __restrict__ X,
                const __bf16* __restrict__ Xbf,   // bf16 X (valid iff use_xbf)
                int use_xbf,
                const float* __restrict__ bfv, const float* __restrict__ biv,
                const float* __restrict__ bgv, const float* __restrict__ bov,
                const __bf16* __restrict__ Wbf,   // [NWG][4][16][1024] bf16
                __bf16* __restrict__ Hbuf,        // 2 ping-pong buffers of [64][512]
                unsigned* __restrict__ bar,
                float* __restrict__ out) {
  extern __shared__ __align__(64) unsigned char smem[];
  __bf16* sW = (__bf16*)smem;               // 65536 halfs
  __bf16* sC = sW + 64 * COMB;              // 65536 halfs
  float*  sG = (float*)(smem + 2 * 64 * COMB * 2);
  float*  sB = sG + 4096;

  const int tid = threadIdx.x;
  const int wg  = blockIdx.x;
  const int n0  = wg * 16;                  // owned hidden slice [n0, n0+16)
  const int lane = tid & 31, wave = tid >> 5;

  // ---- bias staging (f32, small)
  if (tid < 64) {
    int g = tid >> 4, n = tid & 15;
    const float* bp = (g == 0) ? bfv : (g == 1) ? biv : (g == 2) ? bgv : bov;
    sB[tid] = bp[n0 + n];
  }

  // ---- persistent weight slice -> LDS (contiguous 128 KB image)
#ifdef HAVE_TDM
  if (wave == 0) {
    // Tensor Data Mover: one 2D tile, 64 rows x 2048 B, row stride 2048 B.
    unsigned ldsa = (unsigned)(uintptr_t)(void*)sW;         // low 32b = LDS offset
    uint64_t ga = (uint64_t)(uintptr_t)(const void*)(Wbf + (size_t)wg * (4 * 16 * COMB));
    v4u g0 = { 1u,                                           // count=1 (valid D#)
               ldsa,                                         // lds_addr [63:32]
               (unsigned)(ga & 0xFFFFFFFFu),                 // global_addr lo
               (unsigned)((ga >> 32) & 0x01FFFFFFu) | (2u << 30) }; // addr hi | type=2
    v8i g1 = { (int)0x00010000,        // wg_mask=0, data_size=1 (2 bytes)
               (int)(1024u << 16),     // tensor_dim0 = 1024 (bits 79:48 lo half)
               (int)(64u << 16),       // tensor_dim0 hi=0 | tensor_dim1 = 64
               (int)(1024u << 16),     // tensor_dim1 hi=0 | tile_dim0 = 1024
               (int)64,                // tile_dim1 = 64, tile_dim2 = 0
               (int)1024,              // tensor_dim0_stride = 1024 (lo)
               0, 0 };                 // stride hi, tensor_dim1_stride = 0
    v4i gz = { 0, 0, 0, 0 };
#if __clang_major__ >= 23
    v8i gz8 = { 0, 0, 0, 0, 0, 0, 0, 0 };
    __builtin_amdgcn_tensor_load_to_lds(g0, g1, gz, gz, gz8, 0);
#else
    __builtin_amdgcn_tensor_load_to_lds(g0, g1, gz, gz, 0);
#endif
    __builtin_amdgcn_s_wait_tensorcnt(0);
  }
#else
  {
    const unsigned sWoff = (unsigned)(uintptr_t)(void*)sW;
    #pragma unroll
    for (int j = 0; j < 16; ++j) {
      int ci = tid + j * NTHR;              // 8192 chunks of 16 B (128 KB)
      const char* src = (const char*)(Wbf + (size_t)wg * (4 * 16 * COMB)) + ci * 16;
      async_g2l_b128(sWoff + (unsigned)(ci * 16), src);
    }
  }
#endif

  // per-thread persistent cell state: elements e=tid and e=tid+512 of [64][16]
  float cs0 = 0.f, cs1 = 0.f;

  const int mt = wave >> 2, gate = wave & 3;     // one 16x16 tile per wave
  const int colX = lane & 15, sel = lane >> 4;

  const __bf16* arow = sC + (mt * 16 + colX) * COMB;                 // A: row m
  const __bf16* brow = sW + (gate * 16 + colX) * COMB;               // B: col n
  const unsigned sCoff = (unsigned)(uintptr_t)(void*)sC;

  // bias is loop-invariant; make the staging stores visible once, then hoist.
  __syncthreads();
  const float bv = sB[gate * 16 + colX];

  for (int t = 0; t < TSTEPS; ++t) {
    // ---- stage x_t into sC[:, 0:512]
    if (use_xbf) {
      const char* xsrc = (const char*)(Xbf + (size_t)t * (BATCH * INDIM));
      #pragma unroll
      for (int j = 0; j < 8; ++j) {
        int ci = tid + j * NTHR;            // 4096 chunks of 16 B (64 KB)
        int m  = ci >> 6;                   // 64 chunks per 1024-B row
        unsigned lds = sCoff + (unsigned)(m * (COMB * 2) + (ci & 63) * 16);
        async_g2l_b128(lds, xsrc + ci * 16);
      }
    } else {
      const float4* Xt = (const float4*)(X + (size_t)t * (BATCH * INDIM));
      #pragma unroll
      for (int j = 0; j < 16; ++j) {
        int ch = tid + j * NTHR;            // 8192 float4 chunks
        float4 v = Xt[ch];
        int m = ch >> 7;
        int k = (ch & 127) * 4;
        v4bf hv = { (__bf16)v.x, (__bf16)v.y, (__bf16)v.z, (__bf16)v.w };
        *(v4bf*)(sC + m * COMB + k) = hv;
      }
    }
    // ---- stage h (prev step, bf16, L2-hot) into sC[:, 512:1024]
    {
      const char* hsrc = (const char*)(Hbuf + (size_t)(t & 1) * (BATCH * HID));
      #pragma unroll
      for (int j = 0; j < 8; ++j) {
        int ci = tid + j * NTHR;            // 4096 chunks of 16 B (64 KB)
        int m  = ci >> 6;
        unsigned lds = sCoff + (unsigned)(m * (COMB * 2) + INDIM * 2 + (ci & 63) * 16);
        async_g2l_b128(lds, hsrc + ci * 16);
      }
    }
    asm volatile("s_wait_asynccnt 0x0" ::: "memory");
    __syncthreads();

    // ---- GEMM: D(16x16) = comb(16x1024) x W_gate^T(1024x16) + bias
    v8f acc = { bv, bv, bv, bv, bv, bv, bv, bv };
    #pragma unroll 4
    for (int k0 = 0; k0 < COMB; k0 += 32) {
      v8bf alo = *(const v8bf*)(arow + k0 + sel * 8);
      v8bf ahi = *(const v8bf*)(arow + k0 + 16 + sel * 8);
      v16bf a = __builtin_shufflevector(alo, ahi,
                  0,1,2,3,4,5,6,7,8,9,10,11,12,13,14,15);
      v8bf blo = *(const v8bf*)(brow + k0 + sel * 8);
      v8bf bhi = *(const v8bf*)(brow + k0 + 16 + sel * 8);
      v16bf b = __builtin_shufflevector(blo, bhi,
                  0,1,2,3,4,5,6,7,8,9,10,11,12,13,14,15);
      acc = __builtin_amdgcn_wmma_f32_16x16x32_bf16(
                false, a, false, b, (short)0, acc, false, false);
    }
    #pragma unroll
    for (int r = 0; r < 8; ++r)
      sG[gate * 1024 + (mt * 16 + sel * 8 + r) * 16 + colX] = acc[r];
    __syncthreads();

    // ---- elementwise cell update: 2 of the 64x16 owned elements per thread
    __bf16* hdst = Hbuf + (size_t)((t + 1) & 1) * (BATCH * HID);
    float*  outT = out + (size_t)t * (BATCH * HID);
    #pragma unroll
    for (int q = 0; q < 2; ++q) {
      int e = tid + q * NTHR;
      int m = e >> 4, n = e & 15;
      float pf = sG[e];
      float pi = sG[1024 + e];
      float pg = sG[2048 + e];
      float po = sG[3072 + e];
      float fg = sig_f(pf);
      float ig = sig_f(pi);
      float gg = tanh_f(pg);
      float og = sig_f(po);
      float& cs = q ? cs1 : cs0;
      cs = fg * cs + ig * gg;
      float h = og * tanh_f(cs);
      outT[m * HID + n0 + n] = h;
      hdst[m * HID + n0 + n] = (__bf16)h;
      if (t == TSTEPS - 1) {
        out[(size_t)TSTEPS * BATCH * HID + m * HID + n0 + n] = h;                // hx
        out[(size_t)TSTEPS * BATCH * HID + BATCH * HID + m * HID + n0 + n] = cs; // cx
      }
    }

    // ---- device-wide barrier (monotonic counter, no reset races)
    __threadfence();
    __syncthreads();
    if (tid == 0) {
      atomicAdd(bar, 1u);
      unsigned target = (unsigned)(NWG * (t + 1));
      while (__hip_atomic_load(bar, __ATOMIC_RELAXED, __HIP_MEMORY_SCOPE_AGENT) < target)
        __builtin_amdgcn_s_sleep(1);
    }
    __syncthreads();
    __threadfence();
  }
}

extern "C" void kernel_launch(void* const* d_in, const int* in_sizes, int n_in,
                              void* d_out, int out_size, void* d_ws, size_t ws_size,
                              hipStream_t stream) {
  const float* X  = (const float*)d_in[0];
  const float* Wf = (const float*)d_in[1];
  const float* bf = (const float*)d_in[2];
  const float* Wi = (const float*)d_in[3];
  const float* bi = (const float*)d_in[4];
  const float* Wg = (const float*)d_in[5];
  const float* bg = (const float*)d_in[6];
  const float* Wo = (const float*)d_in[7];
  const float* bo = (const float*)d_in[8];

  unsigned char* ws = (unsigned char*)d_ws;
  unsigned* bar  = (unsigned*)ws;
  __bf16*   Wbf  = (__bf16*)(ws + WS_WOFF);
  __bf16*   Hbuf = (__bf16*)(ws + WS_HOFF);
  __bf16*   Xbf  = (__bf16*)(ws + WS_XOFF);

  int use_xbf = (ws_size >= WS_NEED) ? 1 : 0;

  hipMemsetAsync(bar, 0, 64, stream);
  hipMemsetAsync(Hbuf, 0, 2 * BATCH * HID * 2, stream);

  convert_weights_bf16<<<(4 * HID * COMB + 255) / 256, 256, 0, stream>>>(
      Wf, Wi, Wg, Wo, Wbf);
  if (use_xbf) {
    size_t n4 = (size_t)TSTEPS * BATCH * INDIM / 4;
    convert_x_bf16<<<(unsigned)((n4 + 255) / 256), 256, 0, stream>>>(X, Xbf);
  }

  hipFuncSetAttribute((const void*)lstm_persistent,
                      hipFuncAttributeMaxDynamicSharedMemorySize, LDS_BYTES);
  lstm_persistent<<<dim3(NWG), dim3(NTHR), LDS_BYTES, stream>>>(
      X, Xbf, use_xbf, bf, bi, bg, bo, Wbf, Hbuf, bar, (float*)d_out);
}
